// CrossModalAttention_53532472377398
// MI455X (gfx1250) — matrix operand order repaired and checked by reading, not compiled, not run on hardware
//
#include <hip/hip_runtime.h>
#include <hip/hip_bf16.h>

// Problem constants (match reference).
constexpr int Mc = 3;
constexpr int Bc = 65536;
constexpr int Ec = 512;
constexpr int Ac = 256;
constexpr int BT = 64;    // batch rows per block
constexpr int NT = 512;   // threads per block (16 waves)
constexpr int KT = 32;    // K step per WMMA (bf16 16x16x32)
constexpr int XP = 528;   // LDS pitch for x tiles (bf16 elems, 16B-aligned rows)
constexpr int WPI = 40;   // LDS pitch for transposed weight tiles (16B-aligned)

typedef __attribute__((ext_vector_type(16))) __bf16 v16bf;
typedef __attribute__((ext_vector_type(8)))  float  v8f;

union FragAB { v16bf v; uint4 q[2]; unsigned u[8]; };

__device__ __forceinline__ unsigned short f2bf(float x) {
  unsigned int u = __float_as_uint(x);
  u += 0x7FFFu + ((u >> 16) & 1u);   // round-to-nearest-even
  return (unsigned short)(u >> 16);
}

// Async global->LDS copy (CDNA5): 16 bytes per lane, tracked by ASYNCcnt.
// Generic pointers to __shared__ carry the LDS byte address in their low 32 bits.
__device__ __forceinline__ void asyncCopy16(const unsigned short* lds,
                                            const unsigned short* g) {
  unsigned d = (unsigned)(unsigned long long)lds;
  asm volatile("global_load_async_to_lds_b128 %0, %1, off"
               :: "v"(d), "v"(g) : "memory");
}
__device__ __forceinline__ void asyncWait() {
  asm volatile("s_wait_asynccnt 0x0" ::: "memory");
}
// Packed bf16 multiply (VOP3P V_PK_MUL_BF16).
__device__ __forceinline__ unsigned pkmulBf16(unsigned a, unsigned b) {
  unsigned d;
  asm volatile("v_pk_mul_bf16 %0, %1, %2" : "=v"(d) : "v"(a), "v"(b));
  return d;
}

// ---------------------------------------------------------------------------
// Kernel 0: one-time fp32 -> bf16 weight conversion with transpose, so WMMA
// B-fragments become contiguous 32-byte runs:  wT[..][n][e] = W[..][e][n].
// ---------------------------------------------------------------------------
__global__ __launch_bounds__(256) void convert_weights_kernel(
    const float* __restrict__ Wq, const float* __restrict__ Wk,
    const float* __restrict__ Wt,
    unsigned short* __restrict__ wqT, unsigned short* __restrict__ wkT,
    unsigned short* __restrict__ wtT)
{
  const int stride = gridDim.x * blockDim.x;
  const int gid = blockIdx.x * blockDim.x + threadIdx.x;
  const int NQ = Mc * Ac * Ec;          // 393216
  const int NW = Mc * Mc * Ec * Ec;     // 2359296
  for (int i = gid; i < NQ; i += stride) {
    int m = i / (Ac * Ec), r = i % (Ac * Ec);
    int n = r / Ec, e = r % Ec;
    wqT[i] = f2bf(Wq[((size_t)m * Ec + e) * Ac + n]);
    wkT[i] = f2bf(Wk[((size_t)m * Ec + e) * Ac + n]);
  }
  for (int i = gid; i < NW; i += stride) {
    int p = i / (Ec * Ec), r = i % (Ec * Ec);
    int f = r / Ec, e = r % Ec;
    wtT[i] = f2bf(Wt[((size_t)p * Ec + e) * Ec + f]);
  }
}

// ---------------------------------------------------------------------------
// Kernel 1: Q/K projections for all 3 modalities (register accumulators,
// async-staged LDS weight tiles) + tanh scores + masked softmax -> alpha.
// Wave layout: rb = wave>>2 owns rows rb*16..+16; ca = wave&3 owns a-cols
// ca*64..+64 (4 col tiles). All 6 (m,proj) accumulators stay in registers;
// every loop indexing them is fully unrolled so indices are literals.
// ---------------------------------------------------------------------------
__global__ __launch_bounds__(NT) void qk_alpha_kernel(
    const float* __restrict__ x,
    const unsigned short* __restrict__ wqT, const float* __restrict__ bq,
    const unsigned short* __restrict__ wkT, const float* __restrict__ bk,
    const float* __restrict__ vq, float* __restrict__ alpha)
{
  __shared__ unsigned short xk[BT * XP];         // 67.6 KB
  __shared__ unsigned short wl[2][Ac * WPI];     // 2 x 20 KB (double buffer)
  __shared__ float red[9][BT][4];                // 9.2 KB

  const int t = threadIdx.x, lane = t & 31, w = t >> 5;
  const int h = lane >> 4, ln = lane & 15;
  const int rb = w >> 2, ca = w & 3;
  const int b0 = blockIdx.x * BT;

  v8f acc[6][4];
#pragma unroll
  for (int i = 0; i < 6; ++i)
#pragma unroll
    for (int j = 0; j < 4; ++j) acc[i][j] = (v8f){};

#pragma unroll
  for (int m = 0; m < Mc; ++m) {
    __syncthreads();
    // Stage x[m] 64x512 tile as bf16.
    const float* xs = x + ((size_t)m * Bc + b0) * Ec;
#pragma unroll
    for (int i = 0; i < (BT * Ec) / NT; ++i) {
      int idx = i * NT + t;
      int r = idx >> 9, e = idx & 511;
      xk[r * XP + e] = f2bf(xs[(size_t)r * Ec + e]);
    }
    __syncthreads();

#pragma unroll
    for (int proj = 0; proj < 2; ++proj) {
      const unsigned short* WT = (proj ? wkT : wqT) + (size_t)m * Ac * Ec;
      const int aidx = m * 2 + proj;
      // Prime buffer 0 (K-tile 32x256, transposed in LDS).
#pragma unroll
      for (int i = 0; i < 2; ++i) {
        int cc = i * NT + t;                  // 1024 16B chunks
        int n = cc >> 2, ko = (cc & 3) * 8;
        asyncCopy16(&wl[0][n * WPI + ko], WT + (size_t)n * Ec + ko);
      }
      asyncWait(); __syncthreads();

#pragma unroll
      for (int kk = 0; kk < Ec / KT; ++kk) {
        int cur = kk & 1;
        if (kk + 1 < Ec / KT) {
          int k0 = (kk + 1) * KT;
#pragma unroll
          for (int i = 0; i < 2; ++i) {
            int cc = i * NT + t;
            int n = cc >> 2, ko = (cc & 3) * 8;
            asyncCopy16(&wl[cur ^ 1][n * WPI + ko], WT + (size_t)n * Ec + k0 + ko);
          }
        }
        FragAB a;
        const uint4* xr = (const uint4*)(xk + (rb * 16 + ln) * XP + kk * KT + 16 * h);
        a.q[0] = xr[0];
        a.q[1] = xr[1];
#pragma unroll
        for (int ct = 0; ct < 4; ++ct) {
          int n = ca * 64 + ct * 16 + ln;
          FragAB bf;
          const uint4* wb = (const uint4*)(&wl[cur][n * WPI + 16 * h]);
          bf.q[0] = wb[0]; bf.q[1] = wb[1];
          acc[aidx][ct] = __builtin_amdgcn_wmma_f32_16x16x32_bf16(
              false, a.v, false, bf.v, (short)0, acc[aidx][ct], false, false);
        }
        if (kk + 1 < Ec / KT) { asyncWait(); __syncthreads(); }
      }
    }
  }

  // Scores: lane-local tanh combine (q- and k-projections share positions),
  // shuffle-reduce across 16 lanes, LDS-reduce across the 4 ca waves.
  float bqa[3][4], bka[3][4], vqa[3][4];
#pragma unroll
  for (int q = 0; q < 3; ++q)
#pragma unroll
    for (int ct = 0; ct < 4; ++ct) {
      int a = ca * 64 + ct * 16 + ln;
      bqa[q][ct] = bq[q * Ac + a];
      bka[q][ct] = bk[q * Ac + a];
      vqa[q][ct] = vq[q * Ac + a];
    }
#pragma unroll
  for (int q = 0; q < 3; ++q)
#pragma unroll
    for (int k = 0; k < 3; ++k) {
      float vals[8];
#pragma unroll
      for (int r = 0; r < 8; ++r) {
        float s = 0.f;
#pragma unroll
        for (int ct = 0; ct < 4; ++ct) {
          float qv = acc[q * 2 + 0][ct][r] + bqa[q][ct];
          float kv = acc[k * 2 + 1][ct][r] + bka[k][ct];
          s += vqa[q][ct] * tanhf(qv + kv);
        }
        vals[r] = s;
      }
#pragma unroll
      for (int s2 = 1; s2 < 16; s2 <<= 1)
#pragma unroll
        for (int r = 0; r < 8; ++r) vals[r] += __shfl_xor(vals[r], s2, 32);
      if (ln == 0)
#pragma unroll
        for (int r = 0; r < 8; ++r)
          red[q * 3 + k][rb * 16 + 8 * h + r][ca] = vals[r];
    }
  __syncthreads();

  if (t < BT) {
    float s[9];
#pragma unroll
    for (int i = 0; i < 9; ++i)
      s[i] = red[i][t][0] + red[i][t][1] + red[i][t][2] + red[i][t][3];
#pragma unroll
    for (int q = 0; q < 3; ++q) {
      float mx = -3.4e38f;
#pragma unroll
      for (int k = 0; k < 3; ++k)
        if (k != q) mx = fmaxf(mx, s[q * 3 + k]);
      float e[3], sum = 0.f;
#pragma unroll
      for (int k = 0; k < 3; ++k) {
        e[k] = (k == q) ? 0.f : __expf(s[q * 3 + k] - mx);
        sum += e[k];
      }
      float inv = 1.f / sum;
#pragma unroll
      for (int k = 0; k < 3; ++k)
        alpha[(size_t)(q * 3 + k) * Bc + b0 + t] = e[k] * inv;
    }
  }
}

// ---------------------------------------------------------------------------
// Kernel 2: att[q] = x[q] + sum_k alpha*(x[k] @ Wt[q,k] + bt[q,k]) with
// alpha folded into the bf16 A-fragments (v_pk_mul_bf16, per-lane scalar),
// so one register accumulator per (q, tile) integrates over K and k.
// Residual + bt + concat-LayerNorm applied in registers, written once.
// ---------------------------------------------------------------------------
__global__ __launch_bounds__(NT) void att_ln_kernel(
    const float* __restrict__ x,    const unsigned short* __restrict__ wtT,
    const float* __restrict__ bt,   const float* __restrict__ gamma,
    const float* __restrict__ beta, const float* __restrict__ alpha,
    float* __restrict__ out)
{
  __shared__ unsigned short xk[BT * XP];         // 67.6 KB
  __shared__ unsigned short wl[2][Ec * WPI];     // 2 x 40 KB
  __shared__ float alph[9][BT];                  // 2.3 KB
  __shared__ float red[BT][4][2];                // 2 KB
  __shared__ float stats[BT][2];

  const int t = threadIdx.x, lane = t & 31, w = t >> 5;
  const int h = lane >> 4, ln = lane & 15;
  const int rb = w >> 2, ca = w & 3;
  const int b0 = blockIdx.x * BT;

  for (int i = t; i < 9 * BT; i += NT)
    alph[i / BT][i % BT] = alpha[(size_t)(i / BT) * Bc + b0 + (i % BT)];

  v8f tot[3][8];
#pragma unroll
  for (int q = 0; q < 3; ++q)
#pragma unroll
    for (int j = 0; j < 8; ++j) tot[q][j] = (v8f){};

#pragma unroll
  for (int k = 0; k < Mc; ++k) {
    __syncthreads();
    const float* xs = x + ((size_t)k * Bc + b0) * Ec;
#pragma unroll
    for (int i = 0; i < (BT * Ec) / NT; ++i) {
      int idx = i * NT + t;
      int r = idx >> 9, e = idx & 511;
      xk[r * XP + e] = f2bf(xs[(size_t)r * Ec + e]);
    }
    __syncthreads();

#pragma unroll
    for (int q = 0; q < Mc; ++q) {
      if (q == k) continue;
      const unsigned short* WT = wtT + (size_t)(q * 3 + k) * Ec * Ec;
      // Per-lane alpha (A-fragment row = lane index) as packed bf16.
      unsigned short ab = f2bf(alph[q * 3 + k][rb * 16 + ln]);
      const unsigned ap = ((unsigned)ab << 16) | ab;

      // Prime buffer 0 (K-tile 32x512, transposed in LDS).
#pragma unroll
      for (int i = 0; i < 4; ++i) {
        int cc = i * NT + t;                  // 2048 16B chunks
        int n = cc >> 2, ko = (cc & 3) * 8;
        asyncCopy16(&wl[0][n * WPI + ko], WT + (size_t)n * Ec + ko);
      }
      asyncWait(); __syncthreads();

#pragma unroll
      for (int kk = 0; kk < Ec / KT; ++kk) {
        int cur = kk & 1;
        if (kk + 1 < Ec / KT) {
          int k0 = (kk + 1) * KT;
#pragma unroll
          for (int i = 0; i < 4; ++i) {
            int cc = i * NT + t;
            int n = cc >> 2, ko = (cc & 3) * 8;
            asyncCopy16(&wl[cur ^ 1][n * WPI + ko], WT + (size_t)n * Ec + k0 + ko);
          }
        }
        FragAB a;
        const uint4* xr = (const uint4*)(xk + (rb * 16 + ln) * XP + kk * KT + 16 * h);
        a.q[0] = xr[0];
        a.q[1] = xr[1];
#pragma unroll
        for (int i = 0; i < 8; ++i) a.u[i] = pkmulBf16(a.u[i], ap);
#pragma unroll
        for (int ct = 0; ct < 8; ++ct) {
          int n = (ca * 8 + ct) * 16 + ln;
          FragAB bf;
          const uint4* wb = (const uint4*)(&wl[cur][n * WPI + 16 * h]);
          bf.q[0] = wb[0]; bf.q[1] = wb[1];
          tot[q][ct] = __builtin_amdgcn_wmma_f32_16x16x32_bf16(
              false, a.v, false, bf.v, (short)0, tot[q][ct], false, false);
        }
        if (kk + 1 < Ec / KT) { asyncWait(); __syncthreads(); }
      }
    }
  }

  // Residual + alpha-weighted bt, in registers.
#pragma unroll
  for (int q = 0; q < 3; ++q) {
    const int ksA = (q == 0) ? 1 : 0;
    const int ksB = (q == 2) ? 1 : 2;
#pragma unroll
    for (int ct = 0; ct < 8; ++ct) {
      int n = (ca * 8 + ct) * 16 + ln;
      float btA = bt[(size_t)(q * 3 + ksA) * Ec + n];
      float btB = bt[(size_t)(q * 3 + ksB) * Ec + n];
#pragma unroll
      for (int r = 0; r < 8; ++r) {
        int row = rb * 16 + 8 * h + r;
        tot[q][ct][r] += x[((size_t)q * Bc + b0 + row) * Ec + n]
                       + alph[q * 3 + ksA][row] * btA
                       + alph[q * 3 + ksB][row] * btB;
      }
    }
  }

  // LayerNorm over 1536: lane partials -> shuffle over 16 lanes -> LDS over ca.
  float sm[8], sq[8];
#pragma unroll
  for (int r = 0; r < 8; ++r) { sm[r] = 0.f; sq[r] = 0.f; }
#pragma unroll
  for (int q = 0; q < 3; ++q)
#pragma unroll
    for (int ct = 0; ct < 8; ++ct)
#pragma unroll
      for (int r = 0; r < 8; ++r) {
        float v = tot[q][ct][r];
        sm[r] += v; sq[r] += v * v;
      }
#pragma unroll
  for (int s2 = 1; s2 < 16; s2 <<= 1)
#pragma unroll
    for (int r = 0; r < 8; ++r) {
      sm[r] += __shfl_xor(sm[r], s2, 32);
      sq[r] += __shfl_xor(sq[r], s2, 32);
    }
  if (ln == 0)
#pragma unroll
    for (int r = 0; r < 8; ++r) {
      int row = rb * 16 + 8 * h + r;
      red[row][ca][0] = sm[r];
      red[row][ca][1] = sq[r];
    }
  __syncthreads();
  if (t < BT) {
    float S = 0.f, Q = 0.f;
#pragma unroll
    for (int c2 = 0; c2 < 4; ++c2) { S += red[t][c2][0]; Q += red[t][c2][1]; }
    float mu  = S * (1.f / 1536.f);
    float var = Q * (1.f / 1536.f) - mu * mu;
    stats[t][0] = mu;
    stats[t][1] = rsqrtf(var + 1e-5f);
  }
  __syncthreads();

#pragma unroll
  for (int q = 0; q < 3; ++q)
#pragma unroll
    for (int ct = 0; ct < 8; ++ct) {
      int n = (ca * 8 + ct) * 16 + ln;
      int j = q * Ec + n;
      float g = gamma[j], be = beta[j];
#pragma unroll
      for (int r = 0; r < 8; ++r) {
        int row = rb * 16 + 8 * h + r;
        out[((size_t)(b0 + row)) * (Mc * Ec) + j] =
            (tot[q][ct][r] - stats[row][0]) * stats[row][1] * g + be;
      }
    }
}

// ---------------------------------------------------------------------------
extern "C" void kernel_launch(void* const* d_in, const int* in_sizes, int n_in,
                              void* d_out, int out_size, void* d_ws, size_t ws_size,
                              hipStream_t stream) {
  const float* x     = (const float*)d_in[0];
  const float* Wq    = (const float*)d_in[1];
  const float* bq    = (const float*)d_in[2];
  const float* Wk    = (const float*)d_in[3];
  const float* bk    = (const float*)d_in[4];
  const float* v     = (const float*)d_in[5];
  const float* Wt    = (const float*)d_in[6];
  const float* bt    = (const float*)d_in[7];
  const float* gamma = (const float*)d_in[8];
  const float* beta  = (const float*)d_in[9];
  float* out = (float*)d_out;

  // Workspace layout (bytes):
  //   [0)        alpha : 9*B floats     = 2,359,296
  //   [2359296)  wqT   : 3*256*512 bf16 =   786,432
  //   [3145728)  wkT   : 3*256*512 bf16 =   786,432
  //   [3932160)  wtT   : 9*512*512 bf16 = 4,718,592   (total ~8.65 MB)
  char* ws = (char*)d_ws;
  float*          alpha = (float*)(ws);
  unsigned short* wqT   = (unsigned short*)(ws + 2359296);
  unsigned short* wkT   = (unsigned short*)(ws + 3145728);
  unsigned short* wtT   = (unsigned short*)(ws + 3932160);

  convert_weights_kernel<<<512, 256, 0, stream>>>(Wq, Wk, Wt, wqT, wkT, wtT);
  dim3 grid(Bc / BT);
  qk_alpha_kernel<<<grid, NT, 0, stream>>>(x, wqT, bq, wkT, bk, v, alpha);
  att_ln_kernel<<<grid, NT, 0, stream>>>(x, wtT, bt, gamma, beta, alpha, out);
}
